// DeepSCRI_30322469110292
// MI455X (gfx1250) — matrix operand order repaired and checked by reading, not compiled, not run on hardware
//
#include <hip/hip_runtime.h>
#include <math.h>

#define BB 4
#define NN 2304
#define DD 256
#define NH 8
#define NL 3
#define MM (BB * NN)         // 9216 rows, 576 row-tiles of 16
#define KDROP 345            // int(0.15 * 2304)
#define LN_EPS 1e-5f

typedef __attribute__((ext_vector_type(16))) _Float16 v16h;
typedef __attribute__((ext_vector_type(8)))  _Float16 v8h;
typedef __attribute__((ext_vector_type(8)))  float    v8f;
typedef __attribute__((ext_vector_type(4)))  unsigned int v4u;
typedef __attribute__((ext_vector_type(8)))  int      v8i;
typedef __attribute__((ext_vector_type(4)))  int      v4i;

__device__ __forceinline__ v8f wmma_f16(v16h a, v16h b, v8f c) {
  return __builtin_amdgcn_wmma_f32_16x16x32_f16(false, a, false, b, (short)0, c,
                                                false, false);
}

// A-matrix fragment (16x32 f16), K order matching B fragments:
// lane<16 : row m=lane,    K = 0..15
// lane>=16: row m=lane-16, K = 16..31
__device__ __forceinline__ v16h load_a_frag(const _Float16* row, int lane) {
  int off = (lane & 16) ? 16 : 0;
  return *(const v16h*)(row + off);
}

// B-matrix fragment (32x16 f16): lane<16 holds col n=lane, K=0..15 contiguous;
// lane>=16 holds col n=lane-16, K=16..31. `col` points at the column of B laid
// out contiguously over K (weights are pre-transposed).
__device__ __forceinline__ v16h load_b_frag(const _Float16* col, int lane) {
  int off = (lane & 16) ? 16 : 0;
  return *(const v16h*)(col + off);
}

// ---------------------------------------------------------------------------
// Tensor Data Mover: async 2D tile load (rows x cols f16, row stride in
// elements) from global into LDS. D# built per CDNA5 ISA.
// One wave-level DMA op, tracked by TENSORcnt.
// ---------------------------------------------------------------------------
__device__ __forceinline__ void tdm_load_2d_f16(_Float16* lds,
                                                const _Float16* gsrc,
                                                int rows, int cols, int stride) {
  unsigned long long ga = (unsigned long long)(uintptr_t)gsrc;
  unsigned int la = (unsigned int)(uintptr_t)lds;  // LDS byte offset
  v4u g0;
  g0[0] = 1u;                                           // count=1, user D#
  g0[1] = la;                                           // lds_addr (bytes)
  g0[2] = (unsigned int)ga;                             // global_addr[31:0]
  g0[3] = (unsigned int)((ga >> 32) & 0x1FFFFFFu) | (2u << 30);  // [56:32]|type=2
  v8i g1 = {};
  g1[0] = 0x00010000;            // data_size=1 (2B), workgroup_mask=0
  g1[1] = (cols & 0xFFFF) << 16; // tensor_dim0[15:0]
  g1[2] = (rows & 0xFFFF) << 16; // tensor_dim0[31:16]=0 | tensor_dim1[15:0]
  g1[3] = (cols & 0xFFFF) << 16; // tensor_dim1[31:16]=0 | tile_dim0
  g1[4] = rows & 0xFFFF;         // tile_dim1 (tile_dim2=0)
  g1[5] = stride;                // tensor_dim0_stride[31:0]
  g1[6] = 0;                     // stride[47:32] | tensor_dim1_stride lo
  g1[7] = 0;
  v4i z4 = {};
#if __has_include(<hip/amd_detail/amd_gfx1250_TDM.h>)
  v8i z8 = {};
  __builtin_amdgcn_tensor_load_to_lds(g0, g1, z4, z4, z8, 0);   // therock 6-arg
#else
  __builtin_amdgcn_tensor_load_to_lds(g0, g1, z4, z4, 0);       // ROCm 7.2 5-arg
#endif
}

// ---------------------------------------------------------------------------
// Weight transpose + f32->f16 convert: w (L,K,Nn) -> wt (L,Nn,K)
// ---------------------------------------------------------------------------
__global__ __launch_bounds__(256) void transpose_f16(const float* __restrict__ w,
                                                     _Float16* __restrict__ wt,
                                                     int K, int Nn) {
  size_t per = (size_t)K * Nn;
  size_t tot = (size_t)NL * per;
  size_t idx = (size_t)blockIdx.x * 256 + threadIdx.x;
  if (idx >= tot) return;
  int l = (int)(idx / per);
  size_t r = idx % per;
  int k = (int)(r / Nn), n = (int)(r % Nn);
  wt[(size_t)l * per + (size_t)n * K + k] = (_Float16)w[idx];
}

// ---------------------------------------------------------------------------
// Patch embed: tokens[b,n,d] = sum_{c,p,q} x * pw + pb[d] + pos[n,d]
// ---------------------------------------------------------------------------
__global__ __launch_bounds__(256) void patch_embed(const float* __restrict__ x,
                                                   const float* __restrict__ pw,
                                                   const float* __restrict__ pb,
                                                   const float* __restrict__ pos,
                                                   float* __restrict__ tokens) {
  int tok = blockIdx.x;  // b*NN + n
  int b = tok / NN, n = tok % NN;
  int hc = n / 48, wc = n % 48;
  int d = threadIdx.x;
  float acc = 0.f;
#pragma unroll
  for (int c = 0; c < 3; ++c)
#pragma unroll
    for (int p = 0; p < 2; ++p)
#pragma unroll
      for (int q = 0; q < 2; ++q) {
        float xv = x[(((size_t)b * 3 + c) * 96 + (hc * 2 + p)) * 96 + (wc * 2 + q)];
        float wv = pw[((d * 3 + c) * 2 + p) * 2 + q];
        acc += xv * wv;
      }
  tokens[(size_t)tok * DD + d] = acc + pb[d] + pos[(size_t)n * DD + d];
}

// ---------------------------------------------------------------------------
// Importance MLP -> sigmoid score per token
// ---------------------------------------------------------------------------
__global__ __launch_bounds__(256) void importance(const float* __restrict__ tokens,
                                                  const float* __restrict__ w1,
                                                  const float* __restrict__ b1,
                                                  const float* __restrict__ w2,
                                                  const float* __restrict__ b2,
                                                  float* __restrict__ scores) {
  __shared__ float tok[DD];
  __shared__ float red[DD];
  int t = blockIdx.x, d = threadIdx.x;
  tok[d] = tokens[(size_t)t * DD + d];
  __syncthreads();
  float acc = b1[d];
  for (int k = 0; k < DD; ++k) acc += tok[k] * w1[k * 256 + d];
  float h = fmaxf(acc, 0.f);
  red[d] = h * w2[d];
  __syncthreads();
  for (int s = 128; s > 0; s >>= 1) {
    if (d < s) red[d] += red[d + s];
    __syncthreads();
  }
  if (d == 0) scores[t] = 1.f / (1.f + expf(-(red[0] + b2[0])));
}

// ---------------------------------------------------------------------------
// Bottom-k mask: stable rank (ties broken by lower index) < KDROP => masked
// ---------------------------------------------------------------------------
__global__ __launch_bounds__(256) void bottomk_mask(const float* __restrict__ scores,
                                                    unsigned char* __restrict__ mask) {
  int b = blockIdx.x;
  const float* s = scores + (size_t)b * NN;
  for (int i = threadIdx.x; i < NN; i += 256) {
    float si = s[i];
    int rank = 0;
    for (int j = 0; j < NN; ++j) {
      float sj = s[j];
      rank += (sj < si) || (sj == si && j < i);
    }
    mask[(size_t)b * NN + i] = (rank < KDROP) ? 1 : 0;
  }
}

// ---------------------------------------------------------------------------
// LayerNorm (per token, D=256) -> f16 output
// ---------------------------------------------------------------------------
__global__ __launch_bounds__(256) void ln_f16(const float* __restrict__ x,
                                              const float* __restrict__ g,
                                              const float* __restrict__ bta,
                                              _Float16* __restrict__ out) {
  __shared__ float red[DD];
  int t = blockIdx.x, d = threadIdx.x;
  float v = x[(size_t)t * DD + d];
  red[d] = v;
  __syncthreads();
  for (int s = 128; s > 0; s >>= 1) {
    if (d < s) red[d] += red[d + s];
    __syncthreads();
  }
  float mean = red[0] * (1.f / DD);
  __syncthreads();
  float c = v - mean;
  red[d] = c * c;
  __syncthreads();
  for (int s = 128; s > 0; s >>= 1) {
    if (d < s) red[d] += red[d + s];
    __syncthreads();
  }
  float var = red[0] * (1.f / DD);
  out[(size_t)t * DD + d] = (_Float16)(c * rsqrtf(var + LN_EPS) * g[d] + bta[d]);
}

// ---------------------------------------------------------------------------
// WMMA GEMM: C(M x Nn) = A(M x K, f16) * Wt(Nn x K, f16)^T + bias
// Each wave computes a 16x64 strip (4 N-tiles share one A fragment:
// 2.5 b128 loads per WMMA instead of 4). 8 waves / block.
// mode 0: write f16   mode 1: exact GELU -> f16   mode 2: resid[m,n] += acc
// ---------------------------------------------------------------------------
__global__ __launch_bounds__(256) void gemm_wmma(const _Float16* __restrict__ A,
                                                 const _Float16* __restrict__ Wt,
                                                 const float* __restrict__ bias,
                                                 _Float16* __restrict__ outH,
                                                 float* __restrict__ resid,
                                                 int Mm, int Kk, int Nn, int mode) {
  int lane = threadIdx.x & 31;
  int wave = threadIdx.x >> 5;
  int t = blockIdx.x * 8 + wave;
  int groupsN = Nn >> 6;  // strips of 64 columns
  int total = (Mm >> 4) * groupsN;
  if (t >= total) return;
  int tm = t / groupsN, tg = t % groupsN;
  const _Float16* aRow = A + (size_t)(tm * 16 + (lane & 15)) * Kk;
  const _Float16* bRow = Wt + (size_t)(tg * 64 + (lane & 15)) * Kk;
  size_t bstr = (size_t)16 * Kk;
  v8f acc[4] = {{}, {}, {}, {}};
  for (int kk = 0; kk < Kk; kk += 32) {
    v16h a = load_a_frag(aRow + kk, lane);
#pragma unroll
    for (int ts = 0; ts < 4; ++ts) {
      v16h b = load_b_frag(bRow + ts * bstr + kk, lane);
      acc[ts] = wmma_f16(a, b, acc[ts]);
    }
  }
  int mbase = tm * 16 + ((lane & 16) ? 8 : 0);
#pragma unroll
  for (int ts = 0; ts < 4; ++ts) {
    int n = tg * 64 + ts * 16 + (lane & 15);
    float bv = bias[n];
#pragma unroll
    for (int j = 0; j < 8; ++j) {
      int m = mbase + j;
      float v = acc[ts][j] + bv;
      if (mode == 1) v = 0.5f * v * (1.0f + erff(v * 0.70710678118f));
      if (mode == 2) {
        resid[(size_t)m * Nn + n] += v;
      } else {
        outH[(size_t)m * Nn + n] = (_Float16)v;
      }
    }
  }
}

// ---------------------------------------------------------------------------
// Flash attention: one wave per (b, h, 16-query tile). dk=32 == WMMA K. Keys
// streamed in chunks of 32 with online softmax; masked keys -> -1e30.
// V chunks are staged into LDS by the Tensor Data Mover (async, TENSORcnt).
// qkv layout (B,N,768): Q cols [h*32, h*32+32), K +256, V +512.
// ---------------------------------------------------------------------------
__global__ __launch_bounds__(256) void attn_flash(const _Float16* __restrict__ qkv,
                                                  const unsigned char* __restrict__ mask,
                                                  _Float16* __restrict__ attnout) {
  __shared__ __attribute__((aligned(128))) _Float16 smem[8][16 * 32 + 32 * 32];
  int lane = threadIdx.x & 31;
  int wave = threadIdx.x >> 5;
  _Float16* Pbuf = &smem[wave][0];    // 16 rows x 32 keys
  _Float16* Vbuf = &smem[wave][512];  // 32 keys x 32 dk
  int wid = blockIdx.x * 8 + wave;    // 4608 total
  int qt = wid % (NN / 16);
  int bh = wid / (NN / 16);
  int h = bh % NH, b = bh / NH;
  const float scale = 0.17677669529663687f;  // 1/sqrt(32)

  int qrow = qt * 16 + (lane & 15);
  v16h qa = load_a_frag(qkv + ((size_t)(b * NN + qrow) * 768 + h * 32), lane);

  v8f o0 = {}, o1 = {};
  float mrun[8], lrun[8];
#pragma unroll
  for (int j = 0; j < 8; ++j) { mrun[j] = -1e30f; lrun[j] = 0.f; }
  const unsigned char* mb = mask + (size_t)b * NN;

  for (int kc = 0; kc < NN; kc += 32) {
    // kick off async V tile (32 keys x 32 f16, row stride 768) -> Vbuf
    tdm_load_2d_f16(Vbuf, qkv + ((size_t)(b * NN + kc) * 768 + 512 + h * 32),
                    32, 32, 768);

    int key0 = kc + (lane & 15);
    const _Float16* kp0 = qkv + ((size_t)(b * NN + key0) * 768 + 256 + h * 32);
    v16h bk0 = load_b_frag(kp0, lane);
    v16h bk1 = load_b_frag(kp0 + 16 * 768, lane);
    v8f z = {};
    v8f s0 = wmma_f16(qa, bk0, z);
    v8f s1 = wmma_f16(qa, bk1, z);
    bool m0 = mb[key0] != 0;
    bool m1 = mb[key0 + 16] != 0;
#pragma unroll
    for (int j = 0; j < 8; ++j) {
      float a0 = m0 ? -1e30f : s0[j] * scale;
      float a1 = m1 ? -1e30f : s1[j] * scale;
      float cm = fmaxf(a0, a1);
      cm = fmaxf(cm, __shfl_xor(cm, 1));
      cm = fmaxf(cm, __shfl_xor(cm, 2));
      cm = fmaxf(cm, __shfl_xor(cm, 4));
      cm = fmaxf(cm, __shfl_xor(cm, 8));
      float mnew = fmaxf(mrun[j], cm);
      float p0 = expf(a0 - mnew);
      float p1 = expf(a1 - mnew);
      float rs = p0 + p1;
      rs += __shfl_xor(rs, 1);
      rs += __shfl_xor(rs, 2);
      rs += __shfl_xor(rs, 4);
      rs += __shfl_xor(rs, 8);
      float corr = expf(mrun[j] - mnew);
      lrun[j] = lrun[j] * corr + rs;
      o0[j] *= corr;
      o1[j] *= corr;
      mrun[j] = mnew;
      int m = ((lane & 16) ? 8 : 0) + j;  // D-layout row
      Pbuf[m * 32 + (lane & 15)] = (_Float16)p0;
      Pbuf[m * 32 + 16 + (lane & 15)] = (_Float16)p1;
    }
    asm volatile("s_wait_dscnt 0" ::: "memory");  // Pbuf stores visible
    __builtin_amdgcn_s_wait_tensorcnt(0);         // V tile landed in LDS
    v16h pa = load_a_frag(Pbuf + (lane & 15) * 32, lane);
    int koff = (lane & 16) ? 16 : 0;
    int n = lane & 15;
    v16h bv0, bv1;
#pragma unroll
    for (int i = 0; i < 16; ++i) {
      bv0[i] = Vbuf[(koff + i) * 32 + n];
      bv1[i] = Vbuf[(koff + i) * 32 + 16 + n];
    }
    o0 = wmma_f16(pa, bv0, o0);
    o1 = wmma_f16(pa, bv1, o1);
    asm volatile("s_wait_dscnt 0" ::: "memory");  // gathers done before next TDM/stores
  }
  int nq = lane & 15;
#pragma unroll
  for (int j = 0; j < 8; ++j) {
    int m = qt * 16 + ((lane & 16) ? 8 : 0) + j;
    float inv = 1.0f / lrun[j];
    size_t base = ((size_t)(b * NN + m)) * DD + h * 32;
    attnout[base + nq] = (_Float16)(o0[j] * inv);
    attnout[base + 16 + nq] = (_Float16)(o1[j] * inv);
  }
}

// ---------------------------------------------------------------------------
// Final LN + mean over tokens -> (B, D)
// ---------------------------------------------------------------------------
__global__ __launch_bounds__(256) void final_ln_mean(const float* __restrict__ tokens,
                                                     const float* __restrict__ g,
                                                     const float* __restrict__ bta,
                                                     float* __restrict__ out) {
  __shared__ float red[DD];
  int b = blockIdx.x, d = threadIdx.x;
  float acc = 0.f;
  for (int n = 0; n < NN; ++n) {
    float v = tokens[((size_t)(b * NN + n)) * DD + d];
    red[d] = v;
    __syncthreads();
    for (int s = 128; s > 0; s >>= 1) {
      if (d < s) red[d] += red[d + s];
      __syncthreads();
    }
    float mean = red[0] * (1.f / DD);
    __syncthreads();
    float c = v - mean;
    red[d] = c * c;
    __syncthreads();
    for (int s = 128; s > 0; s >>= 1) {
      if (d < s) red[d] += red[d + s];
      __syncthreads();
    }
    float var = red[0] * (1.f / DD);
    __syncthreads();
    acc += c * rsqrtf(var + LN_EPS) * g[d] + bta[d];
  }
  out[(size_t)b * DD + d] = acc * (1.f / NN);
}

// ---------------------------------------------------------------------------
extern "C" void kernel_launch(void* const* d_in, const int* in_sizes, int n_in,
                              void* d_out, int out_size, void* d_ws, size_t ws_size,
                              hipStream_t stream) {
  const float* x       = (const float*)d_in[0];
  const float* patch_w = (const float*)d_in[1];
  const float* patch_b = (const float*)d_in[2];
  const float* pos     = (const float*)d_in[3];
  const float* imp_w1  = (const float*)d_in[4];
  const float* imp_b1  = (const float*)d_in[5];
  const float* imp_w2  = (const float*)d_in[6];
  const float* imp_b2  = (const float*)d_in[7];
  const float* ln1_g   = (const float*)d_in[8];
  const float* ln1_b   = (const float*)d_in[9];
  const float* qkv_w   = (const float*)d_in[10];
  const float* qkv_b   = (const float*)d_in[11];
  const float* proj_w  = (const float*)d_in[12];
  const float* proj_b  = (const float*)d_in[13];
  const float* ln2_g   = (const float*)d_in[14];
  const float* ln2_b   = (const float*)d_in[15];
  const float* mlp_w1  = (const float*)d_in[16];
  const float* mlp_b1  = (const float*)d_in[17];
  const float* mlp_w2  = (const float*)d_in[18];
  const float* mlp_b2  = (const float*)d_in[19];
  const float* out_g   = (const float*)d_in[20];
  const float* out_b   = (const float*)d_in[21];
  (void)in_sizes; (void)n_in; (void)out_size; (void)ws_size;

  char* ws = (char*)d_ws;
  size_t off = 0;
  auto carve = [&](size_t bytes) -> void* {
    void* p = ws + off;
    off = (off + bytes + 255) & ~(size_t)255;
    return p;
  };
  float*         tokens  = (float*)carve((size_t)MM * DD * 4);
  _Float16*      hln     = (_Float16*)carve((size_t)MM * DD * 2);
  _Float16*      qkvb    = (_Float16*)carve((size_t)MM * 768 * 2);
  _Float16*      attnout = (_Float16*)carve((size_t)MM * DD * 2);
  _Float16*      mlph    = (_Float16*)carve((size_t)MM * 1024 * 2);
  float*         scores  = (float*)carve((size_t)MM * 4);
  unsigned char* maskb   = (unsigned char*)carve((size_t)MM);
  _Float16*      wqkv    = (_Float16*)carve((size_t)NL * 768 * 256 * 2);
  _Float16*      wproj   = (_Float16*)carve((size_t)NL * 256 * 256 * 2);
  _Float16*      wm1     = (_Float16*)carve((size_t)NL * 1024 * 256 * 2);
  _Float16*      wm2     = (_Float16*)carve((size_t)NL * 256 * 1024 * 2);

  auto cdiv = [](size_t a, size_t b) { return (int)((a + b - 1) / b); };

  // weight prep (f32 -> f16, transposed so B-fragments are contiguous rows)
  transpose_f16<<<cdiv((size_t)NL * 256 * 768, 256), 256, 0, stream>>>(qkv_w, wqkv, 256, 768);
  transpose_f16<<<cdiv((size_t)NL * 256 * 256, 256), 256, 0, stream>>>(proj_w, wproj, 256, 256);
  transpose_f16<<<cdiv((size_t)NL * 256 * 1024, 256), 256, 0, stream>>>(mlp_w1, wm1, 256, 1024);
  transpose_f16<<<cdiv((size_t)NL * 1024 * 256, 256), 256, 0, stream>>>(mlp_w2, wm2, 1024, 256);

  patch_embed<<<MM, 256, 0, stream>>>(x, patch_w, patch_b, pos, tokens);
  importance<<<MM, 256, 0, stream>>>(tokens, imp_w1, imp_b1, imp_w2, imp_b2, scores);
  bottomk_mask<<<BB, 256, 0, stream>>>(scores, maskb);

  auto gemm_blocks = [&](int Nn) { return ((MM / 16) * (Nn / 64) + 7) / 8; };

  for (int l = 0; l < NL; ++l) {
    ln_f16<<<MM, 256, 0, stream>>>(tokens, ln1_g + l * 256, ln1_b + l * 256, hln);
    gemm_wmma<<<gemm_blocks(768), 256, 0, stream>>>(
        hln, wqkv + (size_t)l * 768 * 256, qkv_b + l * 768, qkvb, nullptr,
        MM, 256, 768, 0);
    attn_flash<<<(BB * NH * (NN / 16)) / 8, 256, 0, stream>>>(qkvb, maskb, attnout);
    gemm_wmma<<<gemm_blocks(256), 256, 0, stream>>>(
        attnout, wproj + (size_t)l * 256 * 256, proj_b + l * 256, nullptr, tokens,
        MM, 256, 256, 2);
    ln_f16<<<MM, 256, 0, stream>>>(tokens, ln2_g + l * 256, ln2_b + l * 256, hln);
    gemm_wmma<<<gemm_blocks(1024), 256, 0, stream>>>(
        hln, wm1 + (size_t)l * 1024 * 256, mlp_b1 + l * 1024, mlph, nullptr,
        MM, 256, 1024, 1);
    gemm_wmma<<<gemm_blocks(256), 256, 0, stream>>>(
        mlph, wm2 + (size_t)l * 256 * 1024, mlp_b2 + l * 256, nullptr, tokens,
        MM, 1024, 256, 2);
  }
  final_ln_mean<<<BB, 256, 0, stream>>>(tokens, out_g, out_b, (float*)d_out);
}
